// TypeAwareEdgeUpdate_24223615550200
// MI455X (gfx1250) — matrix operand order repaired and checked by reading, not compiled, not run on hardware
//
#include <hip/hip_runtime.h>
#include <hip/hip_bf16.h>

// ---------------------------------------------------------------------------
// TypeAwareEdgeUpdate for MI455X (gfx1250, wave32, WMMA)
//
//   tmp1 = relu-deferred per-type GEMM  nodes_1 @ W1[type1] + b1[type1]
//   tmp2 = same for stream 2
//   e    = edges @ We + be
//   out  = relu( relu([tmp1,tmp2,e]) @ Wout + bout )
//
// Strategy: grouped GEMM via device-side binning by type (16-aligned buckets,
// -1 padding), bf16x3 split-precision WMMA (fp32-equivalent accuracy),
// weights pre-packed into B-fragment lane layout, edge GEMM fused into the
// final K=768 GEMM through LDS.
// ---------------------------------------------------------------------------

typedef __attribute__((ext_vector_type(16))) __bf16 v16bf;
typedef __attribute__((ext_vector_type(8)))  float  v8f;

#define D_NODE 256
#define D_EDGE 128
#define D_OUTC 256
#define LDST   264   // LDS row stride in floats (keeps 16B alignment, staggers banks)

// --- bf16x3: split 8 consecutive f32 into hi/lo bf16 A-fragment elements ----
// A (16x32 bf16) fragment element e of lane L (half = L>=16):
//   K = (e % 8) + 8*half + 16*(e / 8)   -> two contiguous 8-float chunks.
template <int EBASE>
__device__ __forceinline__ void split8(const float* s, v16bf& hi, v16bf& lo) {
  const float4* p = (const float4*)s;
  float4 u = p[0];
  float4 v = p[1];
  float xs[8] = {u.x, u.y, u.z, u.w, v.x, v.y, v.z, v.w};
#pragma unroll
  for (int j = 0; j < 8; ++j) {
    float  x = xs[j];
    __bf16 h = (__bf16)x;
    hi[EBASE + j] = h;
    lo[EBASE + j] = (__bf16)(x - (float)h);
  }
}

// acc += (ahi+alo) * (bhi+blo), dropping the lo*lo term (~2^-18 relative)
__device__ __forceinline__ v8f mm3(v16bf ahi, v16bf alo, v16bf bhi, v16bf blo, v8f c) {
  c = __builtin_amdgcn_wmma_f32_16x16x32_bf16(false, ahi, false, bhi, (short)0, c, false, false);
  c = __builtin_amdgcn_wmma_f32_16x16x32_bf16(false, ahi, false, blo, (short)0, c, false, false);
  c = __builtin_amdgcn_wmma_f32_16x16x32_bf16(false, alo, false, bhi, (short)0, c, false, false);
  return c;
}

// One GEMM pass: KTN k-tiles of 32 from `arow`, 4 column tiles per wave.
// Packed-B fragment fi = nt*KT_TOTAL + ktOff + kt, 32 lanes x 16 bf16 each.
template <int KTN>
__device__ __forceinline__ void gemm_pass(const float* arow,
                                          const __bf16* whi, const __bf16* wlo,
                                          int ntBase, int ktOff, int ktTotal,
                                          int lane, int half, v8f (&acc)[4]) {
  for (int kt = 0; kt < KTN; ++kt) {
    v16bf ahi{}, alo{};
    const float* s = arow + kt * 32 + half * 8;
    split8<0>(s, ahi, alo);
    split8<8>(s + 16, ahi, alo);
#pragma unroll
    for (int ct = 0; ct < 4; ++ct) {
      const size_t fo =
          (((size_t)(ntBase + ct) * ktTotal + (ktOff + kt)) * 32 + lane) * 16;
      v16bf bhi = *(const v16bf*)(whi + fo);
      v16bf blo = *(const v16bf*)(wlo + fo);
      acc[ct] = mm3(ahi, alo, bhi, blo, acc[ct]);
    }
  }
}

// --- weight packing: W[K,256] row-major (matCount matrices) -> B fragments --
// B (32x16 bf16) fragment element e of lane L: col n = 16*nt + (L%16),
//   K = 32*kt + 8*(L/16) + (e%8) + 16*(e/8).  Fragment fi = (t*16+nt)*KT + kt.
__global__ void k_pack(const float* __restrict__ W, unsigned short* __restrict__ hi_,
                       unsigned short* __restrict__ lo_, int KT, int total) {
  int gid = blockIdx.x * blockDim.x + threadIdx.x;
  if (gid >= total) return;
  __bf16* hi = (__bf16*)hi_;
  __bf16* lo = (__bf16*)lo_;
  int lane = gid & 31;
  int fi   = gid >> 5;
  int kt   = fi % KT;
  int rest = fi / KT;
  int nt   = rest & 15;
  int t    = rest >> 4;
  const int n     = nt * 16 + (lane & 15);
  const int kbase = kt * 32 + (lane >> 4) * 8;
  const float* Wm = W + (size_t)t * (size_t)(KT * 32) * 256;
  __bf16* ph = hi + (size_t)gid * 16;
  __bf16* pl = lo + (size_t)gid * 16;
#pragma unroll
  for (int e = 0; e < 16; ++e) {
    float  x = Wm[(size_t)(kbase + (e & 7) + ((e >> 3) << 4)) * 256 + n];
    __bf16 h = (__bf16)x;
    ph[e] = h;
    pl[e] = (__bf16)(x - (float)h);
  }
}

// --- binning ---------------------------------------------------------------
__global__ void k_count(const int* __restrict__ t1, const int* __restrict__ t2,
                        int* c1, int* c2, int N) {
  int i = blockIdx.x * blockDim.x + threadIdx.x;
  if (i < N) {
    atomicAdd(&c1[t1[i]], 1);
    atomicAdd(&c2[t2[i]], 1);
  }
}

__global__ void k_offsets(const int* __restrict__ c1, const int* __restrict__ c2,
                          int* s1, int* s2, int J) {
  if (threadIdx.x == 0 && blockIdx.x == 0) {
    int a = 0, b = 0;
    for (int t = 0; t < J; ++t) {
      s1[t] = a; a = (a + c1[t] + 15) & ~15;
      s2[t] = b; b = (b + c2[t] + 15) & ~15;
    }
  }
}

__global__ void k_scatter(const int* __restrict__ t1, const int* __restrict__ t2,
                          const int* __restrict__ s1, const int* __restrict__ s2,
                          int* cur1, int* cur2, int* p1, int* p2, int N) {
  int i = blockIdx.x * blockDim.x + threadIdx.x;
  if (i < N) {
    int a = t1[i];
    p1[s1[a] + atomicAdd(&cur1[a], 1)] = i;
    int b = t2[i];
    p2[s2[b] + atomicAdd(&cur2[b], 1)] = i;
  }
}

// --- grouped per-type GEMM: tile = 16 same-type rows, block = 4 waves ------
__global__ __launch_bounds__(128) void gemm_pertype(
    const float* __restrict__ nodes, const int* __restrict__ types,
    const int* __restrict__ perm, const unsigned short* __restrict__ whi_,
    const unsigned short* __restrict__ wlo_, const float* __restrict__ bias,
    float* __restrict__ tmp) {
  const int base = blockIdx.x * 16;
  const int r0   = perm[base];
  if (r0 < 0) return;  // tile fully in alignment padding
  const int t    = types[r0];
  const int lane = threadIdx.x & 31;
  const int wave = threadIdx.x >> 5;
  const int half = lane >> 4;
  const int m    = lane & 15;
  const int myRow = perm[base + m];                 // -1 => padding row
  const float* arow = nodes + (size_t)(myRow < 0 ? r0 : myRow) * D_NODE;

  const __bf16* whi = (const __bf16*)whi_ + (size_t)t * D_NODE * D_OUTC;
  const __bf16* wlo = (const __bf16*)wlo_ + (size_t)t * D_NODE * D_OUTC;

  v8f acc[4] = {};
  gemm_pass<8>(arow, whi, wlo, wave * 4, 0, 8, lane, half, acc);

#pragma unroll
  for (int ct = 0; ct < 4; ++ct) {
    const int   n  = (wave * 4 + ct) * 16 + m;
    const float bv = bias[t * D_OUTC + n];
#pragma unroll
    for (int r = 0; r < 8; ++r) {
      const int M   = r + 8 * half;                 // C/D layout: M = r + 8*(lane/16)
      const int row = __shfl(myRow, M, 32);
      if (row >= 0)
        tmp[(size_t)row * D_OUTC + n] = fmaxf(acc[ct][r] + bv, 0.f);  // store relu(tmp)
    }
  }
}

// --- final fused kernel: edge GEMM -> LDS, then K=768 GEMM vs Wout ---------
__global__ __launch_bounds__(128) void k_final(
    const float* __restrict__ edges, const unsigned short* __restrict__ wehi_,
    const unsigned short* __restrict__ welo_, const float* __restrict__ be,
    const float* __restrict__ tmp1, const float* __restrict__ tmp2,
    const unsigned short* __restrict__ wohi_, const unsigned short* __restrict__ wolo_,
    const float* __restrict__ bout, float* __restrict__ out, int N) {
  __shared__ float eLds[16 * LDST];
  const __bf16* wehi = (const __bf16*)wehi_;
  const __bf16* welo = (const __bf16*)welo_;
  const __bf16* wohi = (const __bf16*)wohi_;
  const __bf16* wolo = (const __bf16*)wolo_;

  const int lane = threadIdx.x & 31;
  const int wave = threadIdx.x >> 5;
  const int half = lane >> 4;
  const int m    = lane & 15;
  const int rowBase = blockIdx.x * 16;
  const int myRow   = rowBase + m;
  const int ldRow   = myRow < N ? myRow : 0;

  // Phase 1: e = relu(edges @ We + be) -> LDS (fp32)
  v8f eacc[4] = {};
  gemm_pass<4>(edges + (size_t)ldRow * D_EDGE, wehi, welo, wave * 4, 0, 4, lane, half, eacc);
#pragma unroll
  for (int ct = 0; ct < 4; ++ct) {
    const int   n  = (wave * 4 + ct) * 16 + m;
    const float bv = be[n];
#pragma unroll
    for (int r = 0; r < 8; ++r)
      eLds[(r + 8 * half) * LDST + n] = fmaxf(eacc[ct][r] + bv, 0.f);
  }
  __syncthreads();

  // Phase 2: out = relu( [tmp1 | tmp2 | e] @ Wout + bout ), K = 768 (24 k-tiles)
  v8f o[4] = {};
  gemm_pass<8>(tmp1 + (size_t)ldRow * D_OUTC, wohi, wolo, wave * 4, 0,  24, lane, half, o);
  gemm_pass<8>(tmp2 + (size_t)ldRow * D_OUTC, wohi, wolo, wave * 4, 8,  24, lane, half, o);
  gemm_pass<8>(&eLds[m * LDST],               wohi, wolo, wave * 4, 16, 24, lane, half, o);

#pragma unroll
  for (int ct = 0; ct < 4; ++ct) {
    const int   n  = (wave * 4 + ct) * 16 + m;
    const float bv = bout[n];
#pragma unroll
    for (int r = 0; r < 8; ++r) {
      const int row = rowBase + r + 8 * half;
      if (row < N) out[(size_t)row * D_OUTC + n] = fmaxf(o[ct][r] + bv, 0.f);
    }
  }
}

// ---------------------------------------------------------------------------
extern "C" void kernel_launch(void* const* d_in, const int* in_sizes, int n_in,
                              void* d_out, int out_size, void* d_ws, size_t ws_size,
                              hipStream_t stream) {
  const float* nodes1 = (const float*)d_in[0];
  const float* nodes2 = (const float*)d_in[1];
  const float* edges  = (const float*)d_in[2];
  const int*   types1 = (const int*)d_in[3];
  const int*   types2 = (const int*)d_in[4];
  const float* W1     = (const float*)d_in[5];
  const float* b1     = (const float*)d_in[6];
  const float* W2     = (const float*)d_in[7];
  const float* b2     = (const float*)d_in[8];
  const float* We     = (const float*)d_in[9];
  const float* be     = (const float*)d_in[10];
  const float* Wout   = (const float*)d_in[11];
  const float* bout   = (const float*)d_in[12];

  const int N    = in_sizes[3];
  const int J    = in_sizes[6] / D_OUTC;              // 17
  const int Npad = (N + J * 16 + 15) & ~15;           // 16-aligned bucket padding

  // workspace carve-up (256B-aligned chunks)
  char*  ws  = (char*)d_ws;
  size_t off = 0;
  auto alloc = [&](size_t bytes) -> void* {
    void* p = ws + off;
    off = (off + bytes + 255) & ~(size_t)255;
    return p;
  };
  float* tmp1  = (float*)alloc((size_t)N * D_OUTC * 4);
  float* tmp2  = (float*)alloc((size_t)N * D_OUTC * 4);
  int*   perms = (int*)alloc((size_t)Npad * 8);       // perm1 | perm2
  int*   perm1 = perms;
  int*   perm2 = perms + Npad;
  int*   ctr   = (int*)alloc(6 * 32 * sizeof(int));
  int *c1 = ctr, *c2 = ctr + 32, *s1 = ctr + 64, *s2 = ctr + 96,
      *u1 = ctr + 128, *u2 = ctr + 160;
  unsigned short* w1hi = (unsigned short*)alloc((size_t)J * D_NODE * D_OUTC * 2);
  unsigned short* w1lo = (unsigned short*)alloc((size_t)J * D_NODE * D_OUTC * 2);
  unsigned short* w2hi = (unsigned short*)alloc((size_t)J * D_NODE * D_OUTC * 2);
  unsigned short* w2lo = (unsigned short*)alloc((size_t)J * D_NODE * D_OUTC * 2);
  unsigned short* wohi = (unsigned short*)alloc((size_t)3 * D_OUTC * D_OUTC * 2);
  unsigned short* wolo = (unsigned short*)alloc((size_t)3 * D_OUTC * D_OUTC * 2);
  unsigned short* wehi = (unsigned short*)alloc((size_t)D_EDGE * D_OUTC * 2);
  unsigned short* welo = (unsigned short*)alloc((size_t)D_EDGE * D_OUTC * 2);

  hipMemsetAsync(ctr, 0, 6 * 32 * sizeof(int), stream);
  hipMemsetAsync(perms, 0xFF, (size_t)Npad * 8, stream);   // perm = -1 everywhere

  // pack weights (hi/lo bf16 split) into B-fragment layout
  {
    int tot = J * 16 * 8 * 32;
    k_pack<<<(tot + 255) / 256, 256, 0, stream>>>(W1, w1hi, w1lo, 8, tot);
    k_pack<<<(tot + 255) / 256, 256, 0, stream>>>(W2, w2hi, w2lo, 8, tot);
  }
  {
    int tot = 16 * 24 * 32;
    k_pack<<<(tot + 255) / 256, 256, 0, stream>>>(Wout, wohi, wolo, 24, tot);
  }
  {
    int tot = 16 * 4 * 32;
    k_pack<<<(tot + 255) / 256, 256, 0, stream>>>(We, wehi, welo, 4, tot);
  }

  // bin rows by type
  k_count<<<(N + 255) / 256, 256, 0, stream>>>(types1, types2, c1, c2, N);
  k_offsets<<<1, 32, 0, stream>>>(c1, c2, s1, s2, J);
  k_scatter<<<(N + 255) / 256, 256, 0, stream>>>(types1, types2, s1, s2, u1, u2,
                                                 perm1, perm2, N);

  // grouped per-type GEMMs
  const int tiles = Npad / 16;
  gemm_pertype<<<tiles, 128, 0, stream>>>(nodes1, types1, perm1, w1hi, w1lo, b1, tmp1);
  gemm_pertype<<<tiles, 128, 0, stream>>>(nodes2, types2, perm2, w2hi, w2lo, b2, tmp2);

  // fused edge GEMM + output GEMM
  k_final<<<(N + 15) / 16, 128, 0, stream>>>(edges, wehi, welo, be, tmp1, tmp2,
                                             wohi, wolo, bout, (float*)d_out, N);
}
